// Model_23124103921945
// MI455X (gfx1250) — matrix operand order repaired and checked by reading, not compiled, not run on hardware
//
#include <hip/hip_runtime.h>
#include <cstdint>
#include <cstddef>

// ---------------------------------------------------------------------------
// 4-layer LSTM (B=128,T=128,D=H=1024) + FC(64) for MI455X / gfx1250.
// bf16 WMMA (v_wmma_f32_16x16x32_bf16) with fp32 accumulation.
// Weights converted to bf16 once per call (67 MB total -> L2-resident, 192MB).
// Recurrent step fuses [x_t ; h_{t-1}] @ [W_ih ; W_hh]^T (K=2048) with the
// gate nonlinearities and the c/h state update in a single kernel per t.
// ---------------------------------------------------------------------------

typedef unsigned short bfraw;  // bf16 bits (avoid __bf16 in host-side code)

typedef __attribute__((ext_vector_type(16))) __bf16 v16bf;
typedef __attribute__((ext_vector_type(8)))  __bf16 v8bf;
typedef __attribute__((ext_vector_type(8)))  float  v8f;
typedef __attribute__((ext_vector_type(4)))  unsigned short v4us;

static constexpr int L_ = 4, B_ = 128, T_ = 128, D_ = 1024, H_ = 1024, O_ = 64;
static constexpr int G_ = 4 * H_;  // 4096 gate outputs

// ---------------- helpers ----------------
__device__ __forceinline__ unsigned short f32_to_bf16_bits(float f) {
  unsigned u = __builtin_bit_cast(unsigned, f);
  u += 0x7FFFu + ((u >> 16) & 1u);  // round-to-nearest-even
  return (unsigned short)(u >> 16);
}
__device__ __forceinline__ float bf16_bits_to_f32(unsigned short h) {
  return __builtin_bit_cast(float, (unsigned)h << 16);
}
__device__ __forceinline__ float sigmoidf_(float x) {
  return 1.0f / (1.0f + __expf(-x));
}
__device__ __forceinline__ float tanhf_(float x) {
  x = fminf(15.0f, fmaxf(-15.0f, x));
  float e = __expf(-2.0f * x);
  return (1.0f - e) / (1.0f + e);
}

// A fragment (16x32 bf16, MxK): lane half selects K-offset; per lane the 16
// elements map to K = {k0..k0+15}, contiguous (matches B fragment).
__device__ __forceinline__ v16bf load_frag_a(const bfraw* __restrict__ p, int k0) {
  v8bf lo = *(const v8bf*)(const void*)(p + k0);
  v8bf hi = *(const v8bf*)(const void*)(p + k0 + 8);
  return __builtin_shufflevector(lo, hi, 0, 1, 2, 3, 4, 5, 6, 7,
                                 8, 9, 10, 11, 12, 13, 14, 15);
}
// B fragment (32x16 bf16, KxN): lanes 0-15 hold K=kc*32..+15, lanes 16-31 hold
// K=kc*32+16..+31, contiguous per lane (mirrors ISA sparse-B layout).
__device__ __forceinline__ v16bf load_frag_b(const bfraw* __restrict__ p, int k0) {
  return *(const v16bf*)(const void*)(p + k0);
}

#define WMMA_BF16(A, B, C) \
  __builtin_amdgcn_wmma_f32_16x16x32_bf16(false, (A), false, (B), (short)0, (C), false, false)

// ---------------- conversion kernels ----------------
__global__ void k_cvt_bf16(const float* __restrict__ src, bfraw* __restrict__ dst, int n4) {
  int i = blockIdx.x * blockDim.x + threadIdx.x;
  if (i >= n4) return;
  float4 v = ((const float4*)src)[i];
  v4us o;
  o[0] = f32_to_bf16_bits(v.x);
  o[1] = f32_to_bf16_bits(v.y);
  o[2] = f32_to_bf16_bits(v.z);
  o[3] = f32_to_bf16_bits(v.w);
  ((v4us*)dst)[i] = o;
}

__global__ void k_init_state(const float* __restrict__ h0l, const float* __restrict__ c0l,
                             bfraw* __restrict__ h, float* __restrict__ c, int n) {
  int i = blockIdx.x * blockDim.x + threadIdx.x;
  if (i < n) {
    h[i] = f32_to_bf16_bits(h0l[i]);
    c[i] = c0l[i];
  }
}

// ---------------- fused LSTM timestep ----------------
// gates[128,4096] = [x_t | h] @ [W_ih | W_hh]^T ; then i,f,g,o -> c,h update.
// Grid: 128 blocks = 8 M-tiles x 16 J-groups; block = 128 threads = 4 waves;
// wave w owns output patch (m_tile, j_tile = jg*4+w) and accumulates FOUR
// 16x16 tiles (one per gate) so the nonlinearities fuse locally.
__global__ __launch_bounds__(128) void k_lstm_step(
    const bfraw* __restrict__ x_in,   // [B,T,D] bf16 layer input
    const bfraw* __restrict__ h_in,   // [B,H]  bf16
    const bfraw* __restrict__ Wih,    // [4H,H] bf16
    const bfraw* __restrict__ Whh,    // [4H,H] bf16
    const float* __restrict__ bih,    // [4H]
    const float* __restrict__ bhh,    // [4H]
    float* __restrict__ c,            // [B,H] fp32 in/out (exclusive per wave)
    bfraw* __restrict__ h_out,        // [B,H]  bf16
    bfraw* __restrict__ x_out,        // [B,T,H] bf16 next-layer input
    int t) {
  const int wave  = threadIdx.x >> 5;
  const int lane  = threadIdx.x & 31;
  const int row   = lane & 15;
  const int khalf = lane >> 4;
  const int m0 = (blockIdx.x >> 4) * 16;                  // batch tile base
  const int j0 = ((blockIdx.x & 15) * 4 + wave) * 16;     // hidden-col tile base

  const int b = m0 + row;
  const bfraw* xrow = x_in + (size_t)b * (T_ * D_) + (size_t)t * D_;
  const bfraw* hrow = h_in + (size_t)b * H_;
  const bfraw* wi0 = Wih + (size_t)(0 * H_ + j0 + row) * H_;
  const bfraw* wi1 = Wih + (size_t)(1 * H_ + j0 + row) * H_;
  const bfraw* wi2 = Wih + (size_t)(2 * H_ + j0 + row) * H_;
  const bfraw* wi3 = Wih + (size_t)(3 * H_ + j0 + row) * H_;
  const bfraw* wh0 = Whh + (size_t)(0 * H_ + j0 + row) * H_;
  const bfraw* wh1 = Whh + (size_t)(1 * H_ + j0 + row) * H_;
  const bfraw* wh2 = Whh + (size_t)(2 * H_ + j0 + row) * H_;
  const bfraw* wh3 = Whh + (size_t)(3 * H_ + j0 + row) * H_;

  v8f a0 = {}, a1 = {}, a2 = {}, a3 = {};
  const int ka = khalf * 16;   // A intra-lane K offset
  const int kb = khalf * 16;   // B intra-lane K offset

  // x_t @ W_ih^T   (K = 0..1023)
#pragma unroll 4
  for (int kc = 0; kc < D_ / 32; ++kc) {
    const int k0 = kc * 32;
    v16bf af = load_frag_a(xrow, k0 + ka);
    a0 = WMMA_BF16(af, load_frag_b(wi0, k0 + kb), a0);
    a1 = WMMA_BF16(af, load_frag_b(wi1, k0 + kb), a1);
    a2 = WMMA_BF16(af, load_frag_b(wi2, k0 + kb), a2);
    a3 = WMMA_BF16(af, load_frag_b(wi3, k0 + kb), a3);
  }
  // h_{t-1} @ W_hh^T  (K = 1024..2047)
#pragma unroll 4
  for (int kc = 0; kc < H_ / 32; ++kc) {
    const int k0 = kc * 32;
    v16bf af = load_frag_a(hrow, k0 + ka);
    a0 = WMMA_BF16(af, load_frag_b(wh0, k0 + kb), a0);
    a1 = WMMA_BF16(af, load_frag_b(wh1, k0 + kb), a1);
    a2 = WMMA_BF16(af, load_frag_b(wh2, k0 + kb), a2);
    a3 = WMMA_BF16(af, load_frag_b(wh3, k0 + kb), a3);
  }

  // C/D layout: VGPR r -> M = (khalf ? 8 : 0) + r ; N = j0 + (lane & 15)
  const int j = j0 + row;
  const float bi = bih[j] + bhh[j];
  const float bf = bih[H_ + j] + bhh[H_ + j];
  const float bg = bih[2 * H_ + j] + bhh[2 * H_ + j];
  const float bo = bih[3 * H_ + j] + bhh[3 * H_ + j];
  const int mbase = m0 + khalf * 8;
#pragma unroll
  for (int r = 0; r < 8; ++r) {
    const int m = mbase + r;
    const float iv = sigmoidf_(a0[r] + bi);
    const float fv = sigmoidf_(a1[r] + bf);
    const float gv = tanhf_(a2[r] + bg);
    const float ov = sigmoidf_(a3[r] + bo);
    const size_t ci = (size_t)m * H_ + j;
    const float cn = fv * c[ci] + iv * gv;
    c[ci] = cn;
    const unsigned short hb = f32_to_bf16_bits(ov * tanhf_(cn));
    h_out[ci] = hb;
    x_out[(size_t)m * (T_ * H_) + (size_t)t * H_ + j] = hb;
  }
}

// ---------------- final FC (tiny: 128x64, K=1024, fp32) ----------------
__global__ void k_fc(const bfraw* __restrict__ h, const float* __restrict__ W,
                     const float* __restrict__ bias, float* __restrict__ out) {
  int idx = blockIdx.x * blockDim.x + threadIdx.x;
  if (idx >= B_ * O_) return;
  int b = idx >> 6, o = idx & (O_ - 1);
  const bfraw* hr = h + (size_t)b * H_;
  const float* wr = W + (size_t)o * H_;
  float acc = bias[o];
#pragma unroll 8
  for (int k = 0; k < H_; ++k) acc += bf16_bits_to_f32(hr[k]) * wr[k];
  out[idx] = acc;
}

// ---------------- launcher ----------------
extern "C" void kernel_launch(void* const* d_in, const int* in_sizes, int n_in,
                              void* d_out, int out_size, void* d_ws, size_t ws_size,
                              hipStream_t stream) {
  (void)in_sizes; (void)n_in; (void)out_size; (void)ws_size;
  const float* batch = (const float*)d_in[0];
  const float* h0    = (const float*)d_in[1];
  const float* c0    = (const float*)d_in[2];
  const float* W_ih  = (const float*)d_in[3];
  const float* W_hh  = (const float*)d_in[4];
  const float* b_ih  = (const float*)d_in[5];
  const float* b_hh  = (const float*)d_in[6];
  const float* W_fc  = (const float*)d_in[7];
  const float* b_fc  = (const float*)d_in[8];
  float* out = (float*)d_out;

  char* ws = (char*)d_ws;
  size_t off = 0;
  auto carve = [&](size_t bytes) -> void* {
    void* p = ws + off;
    off += (bytes + 255) & ~(size_t)255;
    return p;
  };
  const size_t wsz = (size_t)L_ * G_ * H_;       // 16.78M elems per weight set
  const size_t xsz = (size_t)B_ * T_ * D_;       // 16.78M elems per x buffer
  bfraw* Wih_bf = (bfraw*)carve(wsz * 2);
  bfraw* Whh_bf = (bfraw*)carve(wsz * 2);
  bfraw* x0     = (bfraw*)carve(xsz * 2);
  bfraw* xA     = (bfraw*)carve(xsz * 2);
  bfraw* xB     = (bfraw*)carve(xsz * 2);
  bfraw* hbuf0  = (bfraw*)carve((size_t)B_ * H_ * 2);
  bfraw* hbuf1  = (bfraw*)carve((size_t)B_ * H_ * 2);
  float* cbuf   = (float*)carve((size_t)B_ * H_ * 4);

  // one-time (per call) bf16 conversion: weights stay L2-resident afterwards
  {
    const int n4 = (int)(wsz / 4);
    k_cvt_bf16<<<(n4 + 255) / 256, 256, 0, stream>>>(W_ih, Wih_bf, n4);
    k_cvt_bf16<<<(n4 + 255) / 256, 256, 0, stream>>>(W_hh, Whh_bf, n4);
    const int nx4 = (int)(xsz / 4);
    k_cvt_bf16<<<(nx4 + 255) / 256, 256, 0, stream>>>(batch, x0, nx4);
  }

  for (int l = 0; l < L_; ++l) {
    const bfraw* xin = (l == 0) ? x0 : ((l & 1) ? xA : xB);
    bfraw* xout = (l & 1) ? xB : xA;
    const int n = B_ * H_;
    k_init_state<<<(n + 255) / 256, 256, 0, stream>>>(
        h0 + (size_t)l * n, c0 + (size_t)l * n, hbuf0, cbuf, n);
    const bfraw* Wi = Wih_bf + (size_t)l * G_ * H_;
    const bfraw* Wh = Whh_bf + (size_t)l * G_ * H_;
    const float* bi = b_ih + (size_t)l * G_;
    const float* bh = b_hh + (size_t)l * G_;
    for (int t = 0; t < T_; ++t) {
      bfraw* hin  = (t & 1) ? hbuf1 : hbuf0;  // ping-pong: no cross-block races
      bfraw* hout = (t & 1) ? hbuf0 : hbuf1;
      k_lstm_step<<<dim3((B_ / 16) * 16), dim3(128), 0, stream>>>(
          xin, hin, Wi, Wh, bi, bh, cbuf, hout, xout, t);
    }
  }
  // after t=127 the newest h lives in hbuf0 (128 swaps)
  k_fc<<<(B_ * O_ + 255) / 256, 256, 0, stream>>>(hbuf0, W_fc, b_fc, out);
}
